// MaskedMHCA_47914655154608
// MI455X (gfx1250) — matrix operand order repaired and checked by reading, not compiled, not run on hardware
//
#include <hip/hip_runtime.h>
#include <hip/hip_bf16.h>
#include <math.h>

typedef __bf16 bf16;
typedef __attribute__((ext_vector_type(16))) __bf16 v16bf;
typedef __attribute__((ext_vector_type(8)))  float  v8f;
typedef int v4i_ __attribute__((vector_size(16)));

#define B_   2
#define C_   1024
#define T_   2048
#define H_   16
#define CH_  64
#define BT_  (B_ * T_)
#define EPS_ 1e-5f
#define SCALE_ 0.125f  /* 1/sqrt(64) */

// ---------------------------------------------------------------------------
// gfx1250 async global->LDS path (ASYNCcnt-tracked), guarded so the code
// still compiles (sync fallback) if this toolchain lacks the builtin.
// Probe round 2 revealed the builtin signature wants v4i pointers:
//   (int4 addrspace(1)*, int4 addrspace(3)*, imm offset, imm cpol)
// ---------------------------------------------------------------------------
#define HAVE_ASYNC_LDS 0
#if defined(__has_builtin)
#if __has_builtin(__builtin_amdgcn_global_load_async_to_lds_b128)
#undef HAVE_ASYNC_LDS
#define HAVE_ASYNC_LDS 1
#endif
#endif

#if HAVE_ASYNC_LDS
typedef __attribute__((address_space(1))) v4i_ g1v4i;
typedef __attribute__((address_space(3))) v4i_ l3v4i;
#endif

#if defined(__has_builtin)
#if __has_builtin(__builtin_amdgcn_s_wait_asynccnt)
#define WAIT_ASYNC(n) __builtin_amdgcn_s_wait_asynccnt(n)
#endif
#endif
#ifndef WAIT_ASYNC
#define WAIT_ASYNC(n) asm volatile("s_wait_asynccnt %0" ::"i"(n) : "memory")
#endif

static __device__ __forceinline__ void async_copy16(bf16* lds_dst, const bf16* gsrc) {
#if HAVE_ASYNC_LDS
  __builtin_amdgcn_global_load_async_to_lds_b128(
      (g1v4i*)(size_t)gsrc, (l3v4i*)lds_dst, 0, 0);
#else
  *(float4*)lds_dst = *(const float4*)gsrc;
#endif
}

// ---------------------------------------------------------------------------
// WMMA fragment helpers (wave32, 16x16x32 bf16, f32 accumulate)
// A layout (16x32, MxK): lane lo=lane&15 -> row M=lo; hs=lane>>4.
//   elems 0..7  -> K = 8*hs + e ; elems 8..15 -> K = 16 + 8*hs + (e-8)
// B layout (32x16, KxN): lane lo -> col N=lo; same K mapping as A
// C/D layout (16x16 f32): VGPR v -> row M = v + 8*hs; lane lo -> col N=lo
// ---------------------------------------------------------------------------
static __device__ __forceinline__ v16bf load_a_frag(const bf16* base, int ld) {
  int lane = threadIdx.x & 31;
  int lo = lane & 15, hs = lane >> 4;
  const bf16* row = base + (size_t)lo * ld;
  v16bf a;
#pragma unroll
  for (int e = 0; e < 8; ++e) a[e] = row[8 * hs + e];
#pragma unroll
  for (int e = 0; e < 8; ++e) a[8 + e] = row[16 + 8 * hs + e];
  return a;
}

static __device__ __forceinline__ v16bf load_b_frag(const bf16* base, int ld) {
  int lane = threadIdx.x & 31;
  int lo = lane & 15, hs = lane >> 4;
  const bf16* p = base + (size_t)lo * ld;
  v16bf b;
#pragma unroll
  for (int e = 0; e < 8; ++e) b[e] = p[8 * hs + e];
#pragma unroll
  for (int e = 0; e < 8; ++e) b[8 + e] = p[16 + 8 * hs + e];
  return b;
}

static __device__ __forceinline__ v8f zero_v8f() {
  v8f z;
#pragma unroll
  for (int i = 0; i < 8; ++i) z[i] = 0.f;
  return z;
}

static __device__ __forceinline__ v8f wmma_bf16(v16bf a, v16bf b, v8f c) {
  return __builtin_amdgcn_wmma_f32_16x16x32_bf16(
      false, a, false, b, (short)0, c, false, false);
}

// ---------------------------------------------------------------------------
// f32 -> bf16 convert (weights)
// ---------------------------------------------------------------------------
__global__ __launch_bounds__(256) void cvt_bf16_kernel(const float* __restrict__ src,
                                                       bf16* __restrict__ dst, int n) {
  int i = blockIdx.x * 256 + threadIdx.x;
  if (i < n) dst[i] = (bf16)src[i];
}

// ---------------------------------------------------------------------------
// Fused depthwise-conv3 + mask + channel LayerNorm for q/k/v.
// One block per (b,t); 256 threads x 4 channels. Output token-major (BT, C) bf16.
// ---------------------------------------------------------------------------
__global__ __launch_bounds__(256) void ln_dwconv_kernel(
    const float* __restrict__ x, const unsigned char* __restrict__ mask,
    const float* __restrict__ wq_dw, const float* __restrict__ wk_dw,
    const float* __restrict__ wv_dw,
    const float* __restrict__ qlw, const float* __restrict__ qlb,
    const float* __restrict__ klw, const float* __restrict__ klb,
    const float* __restrict__ vlw, const float* __restrict__ vlb,
    bf16* __restrict__ qln, bf16* __restrict__ kln, bf16* __restrict__ vln) {
  __shared__ float red[6][256];
  int bt = blockIdx.x;
  int b = bt >> 11;            // / T_
  int t = bt & (T_ - 1);
  int tid = threadIdx.x;
  float mf = mask[b * T_ + t] ? 1.f : 0.f;

  float yq[4], yk[4], yv[4];
  float sq = 0.f, sq2 = 0.f, sk = 0.f, sk2 = 0.f, sv = 0.f, sv2 = 0.f;
#pragma unroll
  for (int i = 0; i < 4; ++i) {
    int c = tid + 256 * i;
    const float* xc = x + ((size_t)(b * C_ + c)) * T_;
    float xm = (t > 0) ? xc[t - 1] : 0.f;
    float x0 = xc[t];
    float xp = (t < T_ - 1) ? xc[t + 1] : 0.f;
    float q = (xm * wq_dw[c * 3 + 0] + x0 * wq_dw[c * 3 + 1] + xp * wq_dw[c * 3 + 2]) * mf;
    float k = (xm * wk_dw[c * 3 + 0] + x0 * wk_dw[c * 3 + 1] + xp * wk_dw[c * 3 + 2]) * mf;
    float v = (xm * wv_dw[c * 3 + 0] + x0 * wv_dw[c * 3 + 1] + xp * wv_dw[c * 3 + 2]) * mf;
    yq[i] = q; yk[i] = k; yv[i] = v;
    sq += q; sq2 += q * q;
    sk += k; sk2 += k * k;
    sv += v; sv2 += v * v;
  }
  red[0][tid] = sq; red[1][tid] = sq2;
  red[2][tid] = sk; red[3][tid] = sk2;
  red[4][tid] = sv; red[5][tid] = sv2;
  __syncthreads();
  for (int s = 128; s > 0; s >>= 1) {
    if (tid < s) {
#pragma unroll
      for (int r = 0; r < 6; ++r) red[r][tid] += red[r][tid + s];
    }
    __syncthreads();
  }
  const float invC = 1.f / (float)C_;
  float muq = red[0][0] * invC, vrq = red[1][0] * invC - muq * muq;
  float muk = red[2][0] * invC, vrk = red[3][0] * invC - muk * muk;
  float muv = red[4][0] * invC, vrv = red[5][0] * invC - muv * muv;
  float iq = rsqrtf(vrq + EPS_);
  float ik = rsqrtf(vrk + EPS_);
  float iv = rsqrtf(vrv + EPS_);

  bf16* qo = qln + (size_t)bt * C_;
  bf16* ko = kln + (size_t)bt * C_;
  bf16* vo = vln + (size_t)bt * C_;
#pragma unroll
  for (int i = 0; i < 4; ++i) {
    int c = tid + 256 * i;
    qo[c] = (bf16)((yq[i] - muq) * iq * qlw[c] + qlb[c]);
    ko[c] = (bf16)((yk[i] - muk) * ik * klw[c] + klb[c]);
    vo[c] = (bf16)((yv[i] - muv) * iv * vlw[c] + vlb[c]);
  }
}

// ---------------------------------------------------------------------------
// WMMA GEMM: Y[o,n] = sum_c W[o,c] * X[n,c] + bias[o]
//   W: (C_, C_) bf16 row-major;  X: (BT_, C_) bf16 token-major.
// Block = 128 threads (4 waves). Block tile 128x64; each wave a 32x64 strip:
// 2 A-fragments reused across 4 B-fragments -> 8 WMMA per 12 b128 loads.
// mode 0: Q -> (B,H,T,CH), *SCALE   mode 1: K -> (B,H,T,CH)
// mode 2: V -> (B,H,CH,T)           mode 3: proj -> f32 (B,C,T) * mask
// ---------------------------------------------------------------------------
__global__ __launch_bounds__(128) void gemm_wmma_kernel(
    const bf16* __restrict__ W, const bf16* __restrict__ X,
    const float* __restrict__ bias,
    bf16* __restrict__ outT, float* __restrict__ outF,
    const unsigned char* __restrict__ mask, int mode, float scale) {
  int wave = threadIdx.x >> 5;
  int lane = threadIdx.x & 31;
  int lo = lane & 15, hs = lane >> 4;
  int o0 = blockIdx.x * 128 + wave * 32;
  int n0 = blockIdx.y * 64;

  v8f acc[2][4];
#pragma unroll
  for (int r = 0; r < 2; ++r)
#pragma unroll
    for (int j = 0; j < 4; ++j) acc[r][j] = zero_v8f();

  for (int kk = 0; kk < C_; kk += 32) {
    v16bf a0 = load_a_frag(W + (size_t)o0 * C_ + kk, C_);
    v16bf a1 = load_a_frag(W + (size_t)(o0 + 16) * C_ + kk, C_);
#pragma unroll
    for (int j = 0; j < 4; ++j) {
      v16bf bfr = load_b_frag(X + (size_t)(n0 + 16 * j) * C_ + kk, C_);
      acc[0][j] = wmma_bf16(a0, bfr, acc[0][j]);
      acc[1][j] = wmma_bf16(a1, bfr, acc[1][j]);
    }
  }

#pragma unroll
  for (int r = 0; r < 2; ++r)
#pragma unroll
  for (int j = 0; j < 4; ++j) {
    int n = n0 + 16 * j + lo;
    int bb = n >> 11;           // / T_
    int t = n & (T_ - 1);
#pragma unroll
    for (int v = 0; v < 8; ++v) {
      int o = o0 + 16 * r + v + 8 * hs;
      float val = acc[r][j][v] + bias[o];
      if (mode == 0) val *= scale;
      if (mode <= 1) {
        int h = o >> 6, ch = o & 63;
        outT[(((size_t)(bb * H_ + h)) * T_ + t) * CH_ + ch] = (bf16)val;
      } else if (mode == 2) {
        int h = o >> 6, ch = o & 63;
        outT[(((size_t)(bb * H_ + h)) * CH_ + ch) * T_ + t] = (bf16)val;
      } else {
        float mf = mask[bb * T_ + t] ? 1.f : 0.f;
        outF[((size_t)(bb * C_ + o)) * T_ + t] = val * mf;
      }
    }
  }
}

// ---------------------------------------------------------------------------
// Flash attention with double-buffered async global->LDS K/V tile prefetch.
// Per (b,h): 64 query rows per block (4 waves x 16 rows). Q pre-scaled.
// Tile i+1 is fetched via GLOBAL_LOAD_ASYNC_TO_LDS while tile i is consumed;
// s_wait_asynccnt(8) retires the previous tile's 8 per-wave async ops.
// ---------------------------------------------------------------------------
__global__ __launch_bounds__(128) void flash_attn_kernel(
    const bf16* __restrict__ Qg, const bf16* __restrict__ Kg,
    const bf16* __restrict__ Vg, const unsigned char* __restrict__ mask,
    bf16* __restrict__ Og) {
  __shared__ bf16 Klds[2][64 * 64];     // [s][c]
  __shared__ bf16 Vlds[2][64 * 64];     // [c][s]
  __shared__ bf16 Plds[4 * 16 * 64];    // per-wave 16x64

  int tid = threadIdx.x;
  int wave = tid >> 5, lane = tid & 31;
  int lo = lane & 15, hs = lane >> 4;
  int bh = blockIdx.y;
  int b = bh >> 4, h = bh & 15;
  int t0 = blockIdx.x * 64;

  const bf16* Qbh = Qg + (size_t)bh * T_ * CH_;
  const bf16* Kbh = Kg + (size_t)bh * T_ * CH_;
  const bf16* Vbh = Vg + (size_t)bh * CH_ * T_;

  v16bf aq[2];
#pragma unroll
  for (int ks = 0; ks < 2; ++ks)
    aq[ks] = load_a_frag(Qbh + (size_t)(t0 + 16 * wave) * CH_ + 32 * ks, CH_);

  v8f accO[4];
  float m_i[8], l_i[8];
#pragma unroll
  for (int j = 0; j < 4; ++j) accO[j] = zero_v8f();
#pragma unroll
  for (int v = 0; v < 8; ++v) { m_i[v] = -1e30f; l_i[v] = 0.f; }

  // issue K/V tile loads for step s0 into buffer pb (8 async ops per wave)
  auto issue_tile = [&](int s0, int pb) {
    for (int idx = tid; idx < 512; idx += 128) {
      async_copy16(&Klds[pb][idx * 8], Kbh + (size_t)s0 * CH_ + idx * 8);
    }
    for (int idx = tid; idx < 512; idx += 128) {
      int c = idx >> 3, s8 = (idx & 7) * 8;
      async_copy16(&Vlds[pb][c * 64 + s8], Vbh + (size_t)c * T_ + s0 + s8);
    }
  };

  issue_tile(0, 0);
  int cur = 0;
  const int nIter = T_ / 64;
  for (int it = 0; it < nIter; ++it) {
    int s0 = it * 64;
    if (it + 1 < nIter) {
      issue_tile(s0 + 64, cur ^ 1);   // prefetch next tile
      WAIT_ASYNC(8);                  // retire current tile's loads
    } else {
      WAIT_ASYNC(0);
    }
    __syncthreads();
    const bf16* Kc = Klds[cur];
    const bf16* Vc = Vlds[cur];

    // S = (Q*scale)^T K, 16x64 strip per wave
    v8f sacc[4];
#pragma unroll
    for (int j = 0; j < 4; ++j) sacc[j] = zero_v8f();
#pragma unroll
    for (int ks = 0; ks < 2; ++ks)
#pragma unroll
      for (int j = 0; j < 4; ++j) {
        v16bf kb = load_b_frag(Kc + (16 * j) * 64 + 32 * ks, 64);
        sacc[j] = wmma_bf16(aq[ks], kb, sacc[j]);
      }

    // key mask (column s)
#pragma unroll
    for (int j = 0; j < 4; ++j) {
      int s = s0 + 16 * j + lo;
      if (!mask[b * T_ + s]) {
#pragma unroll
        for (int v = 0; v < 8; ++v) sacc[j][v] = -1e30f;
      }
    }

    // online softmax per row (row = v + 8*hs); xor masks 1..8 stay in-half
#pragma unroll
    for (int v = 0; v < 8; ++v) {
      float rm = m_i[v];
#pragma unroll
      for (int j = 0; j < 4; ++j) rm = fmaxf(rm, sacc[j][v]);
#pragma unroll
      for (int off = 1; off < 16; off <<= 1)
        rm = fmaxf(rm, __shfl_xor(rm, off, 32));
      float sc = __expf(m_i[v] - rm);
      m_i[v] = rm;
      l_i[v] *= sc;
      float rs = 0.f;
#pragma unroll
      for (int j = 0; j < 4; ++j) {
        float p = __expf(sacc[j][v] - rm);
        sacc[j][v] = p;
        rs += p;
        accO[j][v] *= sc;
      }
#pragma unroll
      for (int off = 1; off < 16; off <<= 1) rs += __shfl_xor(rs, off, 32);
      l_i[v] += rs;
    }

    // P strip -> per-wave LDS (C-layout scatter), then A-fragment gather
    bf16* Pw = Plds + wave * 16 * 64;
#pragma unroll
    for (int j = 0; j < 4; ++j)
#pragma unroll
      for (int v = 0; v < 8; ++v)
        Pw[(v + 8 * hs) * 64 + 16 * j + lo] = (bf16)sacc[j][v];

#pragma unroll
    for (int ks = 0; ks < 2; ++ks) {
      v16bf pa = load_a_frag(Pw + 32 * ks, 64);
#pragma unroll
      for (int j = 0; j < 4; ++j) {
        v16bf vb = load_b_frag(Vc + (16 * j) * 64 + 32 * ks, 64);
        accO[j] = wmma_bf16(pa, vb, accO[j]);
      }
    }
    __syncthreads();   // all waves done with buf[cur] before it is refilled
    cur ^= 1;
  }

  // normalize and store O token-major (BT, C) for the proj GEMM
#pragma unroll
  for (int v = 0; v < 8; ++v) {
    float inv = (l_i[v] > 0.f) ? 1.f / l_i[v] : 0.f;
    int t = t0 + 16 * wave + v + 8 * hs;
#pragma unroll
    for (int j = 0; j < 4; ++j) {
      Og[((size_t)(b * T_ + t)) * C_ + h * 64 + 16 * j + lo] =
          (bf16)(accO[j][v] * inv);
    }
  }
}

// ---------------------------------------------------------------------------
// qx_mask passthrough (second tuple output)
// ---------------------------------------------------------------------------
__global__ __launch_bounds__(256) void copy_mask_kernel(
    const unsigned char* __restrict__ m, unsigned char* __restrict__ dst, int n) {
  int i = blockIdx.x * 256 + threadIdx.x;
  if (i < n) dst[i] = m[i];
}

// ---------------------------------------------------------------------------
extern "C" void kernel_launch(void* const* d_in, const int* in_sizes, int n_in,
                              void* d_out, int out_size, void* d_ws, size_t ws_size,
                              hipStream_t stream) {
  const float* x = (const float*)d_in[0];
  const unsigned char* mask = (const unsigned char*)d_in[1];
  const float* wq_dw = (const float*)d_in[2];
  const float* wk_dw = (const float*)d_in[3];
  const float* wv_dw = (const float*)d_in[4];
  const float* qlw = (const float*)d_in[5];
  const float* qlb = (const float*)d_in[6];
  const float* klw = (const float*)d_in[7];
  const float* klb = (const float*)d_in[8];
  const float* vlw = (const float*)d_in[9];
  const float* vlb = (const float*)d_in[10];
  const float* wq = (const float*)d_in[11];
  const float* bq = (const float*)d_in[12];
  const float* wk = (const float*)d_in[13];
  const float* bk = (const float*)d_in[14];
  const float* wv = (const float*)d_in[15];
  const float* bv = (const float*)d_in[16];
  const float* wp = (const float*)d_in[17];
  const float* bp = (const float*)d_in[18];
  float* out = (float*)d_out;

  char* ws = (char*)d_ws;
  size_t off = 0;
  auto alloc = [&](size_t bytes) -> char* {
    char* p = ws + off;
    off += (bytes + 255) & ~(size_t)255;
    return p;
  };
  const size_t actBytes = (size_t)BT_ * C_ * sizeof(bf16);  // 8 MiB
  const size_t wBytes = (size_t)C_ * C_ * sizeof(bf16);     // 2 MiB
  bf16* qln = (bf16*)alloc(actBytes);
  bf16* kln = (bf16*)alloc(actBytes);
  bf16* vln = (bf16*)alloc(actBytes);
  bf16* wq_b = (bf16*)alloc(wBytes);
  bf16* wk_b = (bf16*)alloc(wBytes);
  bf16* wv_b = (bf16*)alloc(wBytes);
  bf16* wp_b = (bf16*)alloc(wBytes);
  bf16* Qg = (bf16*)alloc(actBytes);
  bf16* Kg = (bf16*)alloc(actBytes);
  bf16* Vg = (bf16*)alloc(actBytes);
  bf16* Og = qln;  // qln dead after the Q projection; reuse for attention out

  // 1) weights -> bf16
  {
    int n = C_ * C_;
    dim3 g((n + 255) / 256), blk(256);
    cvt_bf16_kernel<<<g, blk, 0, stream>>>(wq, wq_b, n);
    cvt_bf16_kernel<<<g, blk, 0, stream>>>(wk, wk_b, n);
    cvt_bf16_kernel<<<g, blk, 0, stream>>>(wv, wv_b, n);
    cvt_bf16_kernel<<<g, blk, 0, stream>>>(wp, wp_b, n);
  }

  // 2) dwconv3 + mask + LayerNorm (q,k,v fused)
  ln_dwconv_kernel<<<dim3(BT_), dim3(256), 0, stream>>>(
      x, mask, wq_dw, wk_dw, wv_dw, qlw, qlb, klw, klb, vlw, vlb,
      qln, kln, vln);

  // 3) QKV projections (WMMA)
  {
    dim3 g(C_ / 128, BT_ / 64), blk(128);
    gemm_wmma_kernel<<<g, blk, 0, stream>>>(wq_b, qln, bq, Qg, nullptr, mask, 0, SCALE_);
    gemm_wmma_kernel<<<g, blk, 0, stream>>>(wk_b, kln, bk, Kg, nullptr, mask, 1, 1.f);
    gemm_wmma_kernel<<<g, blk, 0, stream>>>(wv_b, vln, bv, Vg, nullptr, mask, 2, 1.f);
  }

  // 4) fused flash attention (WMMA + online softmax + async LDS prefetch)
  flash_attn_kernel<<<dim3(T_ / 64, B_ * H_), dim3(128), 0, stream>>>(
      Qg, Kg, Vg, mask, Og);

  // 5) output projection (WMMA) -> f32 (B,C,T) with mask
  gemm_wmma_kernel<<<dim3(C_ / 128, BT_ / 64), dim3(128), 0, stream>>>(
      wp_b, Og, bp, nullptr, out, mask, 3, 1.f);

  // 6) qx_mask tuple output (if the harness sized d_out for it)
  if (out_size > B_ * C_ * T_) {
    unsigned char* mdst = (unsigned char*)d_out + (size_t)B_ * C_ * T_ * sizeof(float);
    copy_mask_kernel<<<dim3((B_ * T_ + 255) / 256), dim3(256), 0, stream>>>(
        mask, mdst, B_ * T_);
  }
}
